// PerformerForSiamese_28810640622003
// MI455X (gfx1250) — matrix operand order repaired and checked by reading, not compiled, not run on hardware
//
#include <hip/hip_runtime.h>
#include <hip/hip_bf16.h>
#include <math.h>

// ---------------------------------------------------------------------------
// Performer encoder forward for MI455X (gfx1250, wave32, WMMA + TDM).
// All large GEMMs run through v_wmma_f32_16x16x32_bf16 with fp32 accumulate.
// GEMM tile staging uses the Tensor Data Mover (tensor_load_to_lds, TENSORcnt)
// with double-buffered LDS so the DMA of tile i+1 overlaps WMMA on tile i.
// Falls back to global->VGPR->LDS staging if the TDM builtin is absent.
// ---------------------------------------------------------------------------

typedef unsigned short u16;
typedef __attribute__((ext_vector_type(16))) __bf16 v16bf;
typedef __attribute__((ext_vector_type(8)))  float  v8f;
typedef __attribute__((ext_vector_type(4)))  unsigned int v4u;
typedef __attribute__((ext_vector_type(8)))  int v8i;
typedef __attribute__((ext_vector_type(4)))  int v4i;

constexpr int cB = 4, cN = 2048, cD = 1024, cH = 16, cL = 4, cM = 256, cF = 4096, cDH = 64;

union FragBF { v16bf bf; uint4 q[2]; };

__device__ __forceinline__ u16 f2bf(float f) {
    unsigned int u = __float_as_uint(f);
    u += 0x7FFFu + ((u >> 16) & 1u);       // round-to-nearest-even
    return (u16)(u >> 16);
}
__device__ __forceinline__ float bf2f(u16 h) {
    return __uint_as_float(((unsigned int)h) << 16);
}

#if __has_builtin(__builtin_amdgcn_tensor_load_to_lds) && __has_builtin(__builtin_amdgcn_s_wait_tensorcnt)
#define HAVE_TDM 1
#else
#define HAVE_TDM 0
#endif

#if HAVE_TDM
// ---------------------------------------------------------------------------
// TDM 2D tile load: rows x 32 bf16 elements from global (row stride K elems)
// into LDS at ldsAddr with 16B padding every 64B row (-> 40-ushort stride).
// D# packing per CDNA5 ISA.
// ---------------------------------------------------------------------------
__device__ __forceinline__ void tdm_load_tile(const u16* gsrc, unsigned int ldsAddr,
                                              int K, int rows) {
    unsigned long long ga = (unsigned long long)(const void*)gsrc;
    // group0: count=1 | lds_addr[63:32] | global_addr[120:64] | type=2 [127:126]
    v4u g0 = { 1u,
               ldsAddr,
               (unsigned int)ga,
               (unsigned int)((ga >> 32) & 0x01FFFFFFull) | 0x80000000u };
    // group1: data_size=1 (2B), pad_enable, pad_interval=3 (16 DW), pad_amount=3 (4 DW)
    unsigned int dim0 = (unsigned int)K;          // tensor_dim0 (elements)
    unsigned int dim1 = (unsigned int)rows;       // tensor_dim1 (rows)
    unsigned long long s0 = (unsigned long long)K;// tensor_dim0_stride (elements)
    int w0 = (1 << 16) | (1 << 20) | (3 << 22) | (3 << 25);
    int w1 = (int)((dim0 & 0xFFFFu) << 16);                       // dim0 lo16 -> [63:48]
    int w2 = (int)((dim0 >> 16) | ((dim1 & 0xFFFFu) << 16));      // dim0 hi16, dim1 lo16
    int w3 = (int)((dim1 >> 16) | (32u << 16));                   // dim1 hi16, tile_dim0=32
    int w4 = (int)(unsigned int)rows;                             // tile_dim1, tile_dim2=0
    int w5 = (int)(s0 & 0xFFFFFFFFull);                           // stride0 lo32
    int w6 = (int)((s0 >> 32) & 0xFFFFull);                       // stride0 hi16, stride1 lo16=0
    int w7 = 0;                                                   // stride1 hi32
    v8i g1 = { w0, w1, w2, w3, w4, w5, w6, w7 };
    v4i g2 = { 0, 0, 0, 0 };
    v4i g3 = { 0, 0, 0, 0 };
#if defined(__clang_major__) && (__clang_major__ >= 23)
    v8i g4 = { 0, 0, 0, 0, 0, 0, 0, 0 };
    __builtin_amdgcn_tensor_load_to_lds(g0, g1, g2, g3, g4, 0);
#else
    __builtin_amdgcn_tensor_load_to_lds(g0, g1, g2, g3, 0);
#endif
}
#endif

// ---------------------------------------------------------------------------
// fp32 -> bf16 weight conversion with transpose: W[K][N] -> Wt[N][K]
// ---------------------------------------------------------------------------
__global__ void convT_k(const float* __restrict__ W, u16* __restrict__ Wt, int K, int N) {
    int idx = blockIdx.x * 256 + threadIdx.x;
    if (idx >= K * N) return;
    int k = idx / N, n = idx % N;
    Wt[(size_t)n * K + k] = f2bf(W[idx]);
}

// ---------------------------------------------------------------------------
// Embedding: h[b,n,:] = tok_emb[x[b,n],:] + pos_emb[n,:]
// ---------------------------------------------------------------------------
__global__ void embed_k(const int* __restrict__ x, const float* __restrict__ tok,
                        const float* __restrict__ pos, float* __restrict__ h) {
    size_t idx = (size_t)blockIdx.x * 256 + threadIdx.x;    // over B*N*D
    int d = (int)(idx & (cD - 1));
    size_t bn = idx >> 10;
    int n = (int)(bn & (cN - 1));
    h[idx] = tok[(size_t)x[bn] * cD + d] + pos[(size_t)n * cD + d];
}

// ---------------------------------------------------------------------------
// LayerNorm row kernel (D=1024, 256 threads, 4 elems each) -> bf16 z
// ---------------------------------------------------------------------------
__global__ __launch_bounds__(256) void layernorm_k(const float* __restrict__ h,
                                                   const float* __restrict__ g,
                                                   const float* __restrict__ c,
                                                   u16* __restrict__ z) {
    int row = blockIdx.x;
    int tid = threadIdx.x;
    const float* hr = h + (size_t)row * cD;
    float vals[4], s = 0.f, s2 = 0.f;
    #pragma unroll
    for (int i = 0; i < 4; i++) {
        float v = hr[tid + i * 256];
        vals[i] = v; s += v; s2 += v * v;
    }
    __shared__ float r0[256], r1[256];
    r0[tid] = s; r1[tid] = s2;
    __syncthreads();
    for (int off = 128; off > 0; off >>= 1) {
        if (tid < off) { r0[tid] += r0[tid + off]; r1[tid] += r1[tid + off]; }
        __syncthreads();
    }
    float mu = r0[0] * (1.0f / cD);
    float var = r1[0] * (1.0f / cD) - mu * mu;
    float rs = rsqrtf(var + 1e-5f);
    #pragma unroll
    for (int i = 0; i < 4; i++) {
        int d = tid + i * 256;
        z[(size_t)row * cD + d] = f2bf((vals[i] - mu) * rs * g[d] + c[d]);
    }
}

// ---------------------------------------------------------------------------
// Tiled bf16 WMMA GEMM: C[Mr,Nc] = A[Mr,K] * Bt[Nc,K]^T (+bias, epilogue)
// Block tile 128x128x32, 8 waves (4x2), each wave 32x64 = 8 wmma tiles.
// TDM build: double-buffered LDS; DMA of tile i+1 overlaps WMMA on tile i.
// MODE: 0=fp32, 1=bf16, 2=gelu->bf16, 3=fp32+residual, 4=split-heads bf16
// ---------------------------------------------------------------------------
#define TILE_M 128
#define TILE_N 128
#define TILE_K 32
#define LDSS   40   // padded row stride in ushorts (80B; 64B data + 16B pad)

template <int MODE>
__global__ __launch_bounds__(256) void gemm_bf16_wmma(
    const u16* __restrict__ A, const u16* __restrict__ Bt,
    const float* __restrict__ bias, const float* __restrict__ res,
    float* __restrict__ outf, u16* __restrict__ outh,
    int Mr, int Nc, int K)
{
#if HAVE_TDM
    __shared__ u16 As[2][TILE_M * LDSS];
    __shared__ u16 Bs[2][TILE_N * LDSS];
#else
    __shared__ u16 As[1][TILE_M * LDSS];
    __shared__ u16 Bs[1][TILE_N * LDSS];
#endif

    const int tid   = threadIdx.x;
    const int row0  = blockIdx.y * TILE_M;
    const int col0  = blockIdx.x * TILE_N;
    const int wave  = tid >> 5;
    const int lane  = tid & 31;
    const int wm    = wave & 3;        // 0..3 -> 32-row strips
    const int wn    = wave >> 2;       // 0..1 -> 64-col strips
    const int lhalf = lane >> 4;       // 0/1  -> WMMA K-half select
    const int l15   = lane & 15;

    v8f acc[2][4];
    #pragma unroll
    for (int i = 0; i < 2; i++)
        #pragma unroll
        for (int j = 0; j < 4; j++)
            acc[i][j] = (v8f){0.f,0.f,0.f,0.f,0.f,0.f,0.f,0.f};

    const int nk = K / TILE_K;

#if HAVE_TDM
    // Per the ISA aperture rule, low 32 bits of a generic LDS address are the
    // LDS byte offset — used directly as D#.lds_addr.
    const unsigned int ldsA0 = (unsigned int)(size_t)(void*)&As[0][0];
    const unsigned int ldsA1 = (unsigned int)(size_t)(void*)&As[1][0];
    const unsigned int ldsB0 = (unsigned int)(size_t)(void*)&Bs[0][0];
    const unsigned int ldsB1 = (unsigned int)(size_t)(void*)&Bs[1][0];
    const u16* agT = A  + (size_t)row0 * K;
    const u16* bgT = Bt + (size_t)col0 * K;

    // Prologue: wave 0 DMAs A-tile 0, wave 1 DMAs B-tile 0 (uniform descriptors).
    if (tid < 64) {
        tdm_load_tile((tid < 32) ? agT : bgT,
                      (tid < 32) ? ldsA0 : ldsB0, K, TILE_M);
    }
    for (int it = 0; it < nk; ++it) {
        const int cur = it & 1;
        if (tid < 64) {
            if (it + 1 < nk) {
                // Kick off the next tile into the alternate buffer, then wait
                // for <=1 outstanding: TDM ops of a wave complete in order, so
                // tile `it` has landed while tile `it+1` streams in behind the
                // WMMA work below.
                const u16* gsrc = (tid < 32) ? (agT + (size_t)(it + 1) * TILE_K)
                                             : (bgT + (size_t)(it + 1) * TILE_K);
                unsigned int dst = (tid < 32) ? ((it + 1) & 1 ? ldsA1 : ldsA0)
                                              : ((it + 1) & 1 ? ldsB1 : ldsB0);
                tdm_load_tile(gsrc, dst, K, TILE_M);
                __builtin_amdgcn_s_wait_tensorcnt(1);
            } else {
                __builtin_amdgcn_s_wait_tensorcnt(0);
            }
        }
        __syncthreads();
#else
    const int lr = tid >> 1;           // 0..127 (tile row loaded by this thread)
    const int lc = (tid & 1) * 16;     // 0/16   (k sub-column)
    const u16* ag = A  + (size_t)(row0 + lr) * K + lc;
    const u16* bg = Bt + (size_t)(col0 + lr) * K + lc;
    for (int it = 0; it < nk; ++it) {
        const int cur = 0;
        uint4 a0 = *(const uint4*)(ag);
        uint4 a1 = *(const uint4*)(ag + 8);
        uint4 b0 = *(const uint4*)(bg);
        uint4 b1 = *(const uint4*)(bg + 8);
        ag += TILE_K; bg += TILE_K;
        if (it + 1 < nk) {                        // CDNA5 global_prefetch_b8
            __builtin_prefetch(ag, 0, 3);
            __builtin_prefetch(bg, 0, 3);
        }
        *(uint4*)&As[0][lr * LDSS + lc]     = a0;
        *(uint4*)&As[0][lr * LDSS + lc + 8] = a1;
        *(uint4*)&Bs[0][lr * LDSS + lc]     = b0;
        *(uint4*)&Bs[0][lr * LDSS + lc + 8] = b1;
        __syncthreads();
#endif

        const u16* Ab = As[cur];
        const u16* Bb = Bs[cur];
        FragBF fa[2], fb[4];
        #pragma unroll
        for (int i = 0; i < 2; i++) {
            // A 16x32 bf16 layout: lanes 0-15 K={0..7,16..23}; lanes 16-31 +8
            int r = wm * 32 + i * 16 + l15;
            fa[i].q[0] = *(const uint4*)&Ab[r * LDSS + lhalf * 8];
            fa[i].q[1] = *(const uint4*)&Ab[r * LDSS + lhalf * 8 + 16];
        }
        #pragma unroll
        for (int j = 0; j < 4; j++) {
            // B 32x16 bf16 layout: lanes 0-15 K={0..7,16..23}; lanes 16-31 +8
            int r = wn * 64 + j * 16 + l15;
            fb[j].q[0] = *(const uint4*)&Bb[r * LDSS + lhalf * 8];
            fb[j].q[1] = *(const uint4*)&Bb[r * LDSS + lhalf * 8 + 16];
        }
        #pragma unroll
        for (int i = 0; i < 2; i++)
            #pragma unroll
            for (int j = 0; j < 4; j++)
                acc[i][j] = __builtin_amdgcn_wmma_f32_16x16x32_bf16(
                    false, fa[i].bf, false, fb[j].bf,
                    (short)0, acc[i][j], false, false);
        __syncthreads();
    }

    // Epilogue. C/D layout: lanes 0-15 N=l15,M=r ; lanes 16-31 N=l15,M=r+8
    #pragma unroll
    for (int i = 0; i < 2; i++) {
        #pragma unroll
        for (int j = 0; j < 4; j++) {
            #pragma unroll
            for (int r = 0; r < 8; r++) {
                int grow = row0 + wm * 32 + i * 16 + lhalf * 8 + r;
                int gcol = col0 + wn * 64 + j * 16 + l15;
                float v = acc[i][j][r];
                if (bias) v += bias[gcol];
                if constexpr (MODE == 3) v += res[(size_t)grow * Nc + gcol];
                if constexpr (MODE == 2)
                    v = 0.5f * v * (1.0f + erff(v * 0.70710678118654752f));
                if constexpr (MODE == 0 || MODE == 3) {
                    outf[(size_t)grow * Nc + gcol] = v;
                } else if constexpr (MODE == 4) {
                    int b = grow >> 11, n = grow & (cN - 1);
                    int h = gcol >> 6,  dh = gcol & (cDH - 1);
                    outh[(((size_t)(b * cH + h) * cN + n) * cDH) + dh] = f2bf(v);
                } else {
                    outh[(size_t)grow * Nc + gcol] = f2bf(v);
                }
            }
        }
    }
}

// ---------------------------------------------------------------------------
// Performer feature map: dd[m] = dn * <q_row, proj[m,:]>, diag = 0.5*dn^2*|q|^2
// IS_Q: out = ratio*(exp(dd - diag - rowmax(dd)) + eps)   (per-row max)
// !IS_Q: out = dd - diag, rowmax[row] = max_m dd          (global max later)
// ---------------------------------------------------------------------------
template <bool IS_Q>
__global__ __launch_bounds__(256) void perf_phi_k(const u16* __restrict__ data,
                                                  const float* __restrict__ proj,
                                                  float* __restrict__ out,
                                                  float* __restrict__ rowmax) {
    int row = blockIdx.x;                      // over B*H*N
    const u16* dr = data + (size_t)row * cDH;
    __shared__ float q[cDH];
    __shared__ float red[256];
    if (threadIdx.x < cDH) q[threadIdx.x] = bf2f(dr[threadIdx.x]);
    __syncthreads();

    int m = threadIdx.x;                       // 256 == cM
    const float* pr = proj + (size_t)m * cDH;
    float dd = 0.f, sq = 0.f;
    #pragma unroll 8
    for (int d = 0; d < cDH; d++) { dd += q[d] * pr[d]; sq += q[d] * q[d]; }
    const float dn = 0.35355339059327373f;     // 64^-0.25
    dd *= dn;
    float diag = sq * 0.5f * dn * dn;

    red[m] = dd;
    __syncthreads();
    for (int off = 128; off > 0; off >>= 1) {
        if (m < off) red[m] = fmaxf(red[m], red[m + off]);
        __syncthreads();
    }
    float mx = red[0];

    if constexpr (IS_Q) {
        out[(size_t)row * cM + m] = 0.0625f * (expf(dd - diag - mx) + 1e-4f);
    } else {
        out[(size_t)row * cM + m] = dd - diag;
        if (m == 0) rowmax[row] = mx;
    }
}

__global__ void reduce_max_bh_k(const float* __restrict__ rowmax, float* __restrict__ mx) {
    int bh = blockIdx.x;
    __shared__ float red[256];
    float m = -3.0e38f;
    for (int n = threadIdx.x; n < cN; n += 256)
        m = fmaxf(m, rowmax[(size_t)bh * cN + n]);
    red[threadIdx.x] = m;
    __syncthreads();
    for (int off = 128; off > 0; off >>= 1) {
        if (threadIdx.x < off) red[threadIdx.x] = fmaxf(red[threadIdx.x], red[threadIdx.x + off]);
        __syncthreads();
    }
    if (threadIdx.x == 0) mx[bh] = red[0];
}

__global__ void kp_exp_k(float* __restrict__ kp, const float* __restrict__ mx) {
    size_t idx = (size_t)blockIdx.x * 256 + threadIdx.x;   // over B*H*N*M
    int bh = (int)(idx >> 19);                             // N*M = 2^19
    kp[idx] = 0.0625f * (expf(kp[idx] - mx[bh]) + 1e-4f);
}

__global__ void ksum_k(const float* __restrict__ kp, float* __restrict__ ksum) {
    int bh = blockIdx.x, m = threadIdx.x;
    float s = 0.f;
    for (int n = 0; n < cN; n++) s += kp[((size_t)bh * cN + n) * cM + m];
    ksum[(size_t)bh * cM + m] = s;
}

__global__ void dinv_k(const float* __restrict__ qp, const float* __restrict__ ksum,
                       float* __restrict__ dinv) {
    int wid = threadIdx.x >> 5, lane = threadIdx.x & 31;
    int row = blockIdx.x * 8 + wid;            // over B*H*N
    int bh = row >> 11;
    float s = 0.f;
    for (int m = lane; m < cM; m += 32)
        s += qp[(size_t)row * cM + m] * ksum[(size_t)bh * cM + m];
    #pragma unroll
    for (int off = 16; off > 0; off >>= 1) s += __shfl_xor(s, off, 32);
    if (lane == 0) dinv[row] = 1.0f / s;
}

// ctx[bh,m,d] = sum_n kp[bh,n,m] * v[bh,n,d] * mask[b,n]
__global__ __launch_bounds__(256) void ctx_k(const float* __restrict__ kp,
                                             const u16* __restrict__ vb,
                                             const unsigned char* __restrict__ mask,
                                             float* __restrict__ ctx) {
    int bh = blockIdx.y, b = bh >> 4;
    int d = threadIdx.x & 63, mo = threadIdx.x >> 6;
    int m = blockIdx.x * 4 + mo;
    float acc = 0.f;
    for (int n = 0; n < cN; n++) {
        float mk = mask[(size_t)b * cN + n] ? 1.0f : 0.0f;
        acc += kp[((size_t)bh * cN + n) * cM + m] *
               bf2f(vb[((size_t)bh * cN + n) * cDH + d]) * mk;
    }
    ctx[((size_t)bh * cM + m) * cDH + d] = acc;
}

// attn[b,n,h*64+d] = dinv[bh,n] * sum_m qp[bh,n,m] * ctx[bh,m,d]  (merged heads, bf16)
__global__ __launch_bounds__(256) void attn_k(const float* __restrict__ qp,
                                              const float* __restrict__ ctxb,
                                              const float* __restrict__ dinv,
                                              u16* __restrict__ zout) {
    int bh = blockIdx.y, b = bh >> 4, h = bh & 15;
    int d = threadIdx.x & 63, no = threadIdx.x >> 6;
    int n = blockIdx.x * 4 + no;
    float acc = 0.f;
    for (int m = 0; m < cM; m++)
        acc += qp[((size_t)bh * cN + n) * cM + m] * ctxb[((size_t)bh * cM + m) * cDH + d];
    acc *= dinv[(size_t)bh * cN + n];
    zout[((size_t)(b * cN + n)) * cD + h * cDH + d] = f2bf(acc);
}

__global__ void mean_k(const float* __restrict__ h, float* __restrict__ out) {
    int b = blockIdx.y;
    int d = blockIdx.x * 256 + threadIdx.x;
    float s = 0.f;
    for (int n = 0; n < cN; n++) s += h[((size_t)b * cN + n) * cD + d];
    out[(size_t)b * cD + d] = s * (1.0f / cN);
}

// ---------------------------------------------------------------------------
// Host orchestration
// ---------------------------------------------------------------------------
extern "C" void kernel_launch(void* const* d_in, const int* in_sizes, int n_in,
                              void* d_out, int out_size, void* d_ws, size_t ws_size,
                              hipStream_t stream) {
    const int*   x       = (const int*)d_in[0];
    const unsigned char* mask = (const unsigned char*)d_in[1];
    const float* tok_emb = (const float*)d_in[2];
    const float* pos_emb = (const float*)d_in[3];
    const float* ln1_g   = (const float*)d_in[4];
    const float* ln1_b   = (const float*)d_in[5];
    const float* wq      = (const float*)d_in[6];
    const float* bq      = (const float*)d_in[7];
    const float* wk      = (const float*)d_in[8];
    const float* bk      = (const float*)d_in[9];
    const float* wv      = (const float*)d_in[10];
    const float* bv      = (const float*)d_in[11];
    const float* wo      = (const float*)d_in[12];
    const float* bo      = (const float*)d_in[13];
    const float* proj    = (const float*)d_in[14];
    const float* ln2_g   = (const float*)d_in[15];
    const float* ln2_b   = (const float*)d_in[16];
    const float* w1      = (const float*)d_in[17];
    const float* b1      = (const float*)d_in[18];
    const float* w2      = (const float*)d_in[19];
    const float* b2      = (const float*)d_in[20];
    float* out = (float*)d_out;

    char* wsp = (char*)d_ws;
    auto bump = [&](size_t bytes) -> void* {
        void* p = (void*)wsp;
        wsp += (bytes + 255) & ~(size_t)255;
        return p;
    };
    const int rows = cB * cN;                                        // 8192
    float* hbuf   = (float*)bump((size_t)rows * cD * 4);             // 32 MB
    u16*   zbuf   = (u16*)  bump((size_t)rows * cD * 2);             // 16 MB
    u16*   qb     = (u16*)  bump((size_t)cB * cH * cN * cDH * 2);
    u16*   kb     = (u16*)  bump((size_t)cB * cH * cN * cDH * 2);
    u16*   vb     = (u16*)  bump((size_t)cB * cH * cN * cDH * 2);
    float* qp     = (float*)bump((size_t)cB * cH * cN * cM * 4);     // 128 MB
    float* kp     = (float*)bump((size_t)cB * cH * cN * cM * 4);     // 128 MB
    float* rowmax = (float*)bump((size_t)cB * cH * cN * 4);
    float* mxb    = (float*)bump((size_t)cB * cH * 4);
    float* ksum   = (float*)bump((size_t)cB * cH * cM * 4);
    float* dinvb  = (float*)bump((size_t)cB * cH * cN * 4);
    float* ctxb   = (float*)bump((size_t)cB * cH * cM * cDH * 4);
    u16*   act    = (u16*)  bump((size_t)rows * cF * 2);             // 64 MB
    u16*   wqT    = (u16*)  bump((size_t)cD * cD * 2);
    u16*   wkT    = (u16*)  bump((size_t)cD * cD * 2);
    u16*   wvT    = (u16*)  bump((size_t)cD * cD * 2);
    u16*   woT    = (u16*)  bump((size_t)cD * cD * 2);
    u16*   w1T    = (u16*)  bump((size_t)cD * cF * 2);
    u16*   w2T    = (u16*)  bump((size_t)cF * cD * 2);
    (void)ws_size; (void)in_sizes; (void)n_in; (void)out_size;

    embed_k<<<(rows * cD) / 256, 256, 0, stream>>>(x, tok_emb, pos_emb, hbuf);

    const int nDD = (cD * cD + 255) / 256;
    const int nDF = (cD * cF + 255) / 256;
    dim3 gD(cD / TILE_N, rows / TILE_M);       // Nc=1024 GEMMs
    dim3 gF(cF / TILE_N, rows / TILE_M);       // Nc=4096 GEMM

    for (int l = 0; l < cL; l++) {
        const float* wq_l = wq + (size_t)l * cD * cD;
        const float* wk_l = wk + (size_t)l * cD * cD;
        const float* wv_l = wv + (size_t)l * cD * cD;
        const float* wo_l = wo + (size_t)l * cD * cD;
        const float* w1_l = w1 + (size_t)l * cD * cF;
        const float* w2_l = w2 + (size_t)l * cF * cD;
        const float* proj_l = proj + (size_t)l * cM * cDH;

        convT_k<<<nDD, 256, 0, stream>>>(wq_l, wqT, cD, cD);
        convT_k<<<nDD, 256, 0, stream>>>(wk_l, wkT, cD, cD);
        convT_k<<<nDD, 256, 0, stream>>>(wv_l, wvT, cD, cD);
        convT_k<<<nDD, 256, 0, stream>>>(wo_l, woT, cD, cD);
        convT_k<<<nDF, 256, 0, stream>>>(w1_l, w1T, cD, cF);
        convT_k<<<nDF, 256, 0, stream>>>(w2_l, w2T, cF, cD);

        layernorm_k<<<rows, 256, 0, stream>>>(hbuf, ln1_g + (size_t)l * cD,
                                              ln1_b + (size_t)l * cD, zbuf);

        // QKV projections -> split heads (bf16 [B,H,N,DH])
        gemm_bf16_wmma<4><<<gD, 256, 0, stream>>>(zbuf, wqT, bq + (size_t)l * cD,
                                                  nullptr, nullptr, qb, rows, cD, cD);
        gemm_bf16_wmma<4><<<gD, 256, 0, stream>>>(zbuf, wkT, bk + (size_t)l * cD,
                                                  nullptr, nullptr, kb, rows, cD, cD);
        gemm_bf16_wmma<4><<<gD, 256, 0, stream>>>(zbuf, wvT, bv + (size_t)l * cD,
                                                  nullptr, nullptr, vb, rows, cD, cD);

        // Performer feature maps + linear attention
        perf_phi_k<true ><<<cB * cH * cN, 256, 0, stream>>>(qb, proj_l, qp, nullptr);
        perf_phi_k<false><<<cB * cH * cN, 256, 0, stream>>>(kb, proj_l, kp, rowmax);
        reduce_max_bh_k<<<cB * cH, 256, 0, stream>>>(rowmax, mxb);
        kp_exp_k<<<(cB * cH * cN * cM) / 256, 256, 0, stream>>>(kp, mxb);
        ksum_k<<<cB * cH, 256, 0, stream>>>(kp, ksum);
        dinv_k<<<(cB * cH * cN) / 8, 256, 0, stream>>>(qp, ksum, dinvb);
        ctx_k<<<dim3(cM / 4, cB * cH), 256, 0, stream>>>(kp, vb, mask, ctxb);
        attn_k<<<dim3(cN / 4, cB * cH), 256, 0, stream>>>(qp, ctxb, dinvb, zbuf);

        // Output projection + residual (h += attn @ Wo + bo)
        gemm_bf16_wmma<3><<<gD, 256, 0, stream>>>(zbuf, woT, bo + (size_t)l * cD,
                                                  hbuf, hbuf, nullptr, rows, cD, cD);

        // FFN
        layernorm_k<<<rows, 256, 0, stream>>>(hbuf, ln2_g + (size_t)l * cD,
                                              ln2_b + (size_t)l * cD, zbuf);
        gemm_bf16_wmma<2><<<gF, 256, 0, stream>>>(zbuf, w1T, b1 + (size_t)l * cF,
                                                  nullptr, nullptr, act, rows, cF, cD);
        gemm_bf16_wmma<3><<<gD, 256, 0, stream>>>(act, w2T, b2 + (size_t)l * cD,
                                                  hbuf, hbuf, nullptr, rows, cD, cF);
    }

    mean_k<<<dim3(cD / 256, cB), 256, 0, stream>>>(hbuf, out);
}
